// Attention_61263413510528
// MI455X (gfx1250) — matrix operand order repaired and checked by reading, not compiled, not run on hardware
//
#include <hip/hip_runtime.h>
#include <cmath>

// ---- CDNA5 wave32 vector types ----
typedef __attribute__((ext_vector_type(16))) _Float16 v16h;
typedef __attribute__((ext_vector_type(8)))  _Float16 v8h;
typedef __attribute__((ext_vector_type(8)))  float    v8f;

#define B_N 2
#define H_N 16
#define S_N 2048
#define D_N 64
#define WAVES 8
#define QTILE 16
#define QBLOCK (WAVES * QTILE)   // 128 query rows per block
#define LDSK 40                  // padded LDS row (halves) to spread banks

union V16H { v16h v; v8h h[2]; _Float16 e[16]; };

// 16 consecutive fp32 -> 16 f16
__device__ __forceinline__ void cvt16(const float* p, _Float16* dst) {
  const float4* p4 = reinterpret_cast<const float4*>(p);
#pragma unroll
  for (int i = 0; i < 4; ++i) {
    float4 x = p4[i];
    dst[4*i+0] = (_Float16)x.x; dst[4*i+1] = (_Float16)x.y;
    dst[4*i+2] = (_Float16)x.z; dst[4*i+3] = (_Float16)x.w;
  }
}
// 8 consecutive fp32 -> 8 f16
__device__ __forceinline__ void cvt8(const float* p, _Float16* dst) {
  const float4* p4 = reinterpret_cast<const float4*>(p);
#pragma unroll
  for (int i = 0; i < 2; ++i) {
    float4 x = p4[i];
    dst[4*i+0] = (_Float16)x.x; dst[4*i+1] = (_Float16)x.y;
    dst[4*i+2] = (_Float16)x.z; dst[4*i+3] = (_Float16)x.w;
  }
}

// xor-lane exchange via ds_swizzle (group-of-32 mode, and_mask=0x1f)
template<int MASK>
__device__ __forceinline__ float swz_xor(float x) {
  int r = __builtin_amdgcn_ds_swizzle(__builtin_bit_cast(int, x),
                                      (MASK << 10) | 0x1f);
  return __builtin_bit_cast(float, r);
}

__device__ __forceinline__ v8f wmma_f16(v16h a, v16h b, v8f c) {
  // D = A*B + C ; 16x16x32, f16 in, f32 acc
  return __builtin_amdgcn_wmma_f32_16x16x32_f16(
      /*neg_a=*/false, a, /*neg_b=*/false, b,
      /*c_mod=*/(short)0, c, /*reuse_a=*/false, /*reuse_b=*/false);
}

__global__ __launch_bounds__(256)
void fca_attn_kernel(const float* __restrict__ Q,  const float* __restrict__ K,
                     const float* __restrict__ V,  const float* __restrict__ Qi,
                     const float* __restrict__ Ki, const float* __restrict__ Vi,
                     const int* __restrict__ MSK,  float* __restrict__ out)
{
  __shared__ __align__(16) _Float16 ldsP[WAVES][QTILE][LDSK];

  const int tid  = threadIdx.x;
  const int lane = tid & 31;
  const int wave = tid >> 5;
  const int hs   = lane >> 4;   // lane-half select (0/1)
  const int l16  = lane & 15;

  const int nqb  = S_N / QBLOCK;           // 16
  const int qblk = blockIdx.x % nqb;
  const int bh   = blockIdx.x / nqb;       // b*H + h
  const int b    = bh / H_N;

  const int qbaseW = qblk * QBLOCK + wave * QTILE;
  const size_t headOff = (size_t)bh * S_N * D_N;

  // ---- Q / Qi operands in A-layout (two K=32 d-chunks each), fp32->f16 ----
  V16H qa[2], qia[2];
  {
    const float* qp  = Q  + headOff + (size_t)(qbaseW + l16) * D_N;
    const float* qip = Qi + headOff + (size_t)(qbaseW + l16) * D_N;
#pragma unroll
    for (int c = 0; c < 2; ++c) {
      const int s0 = c * 32 + hs * 16;     // A layout: contiguous seg {s0..s0+15}
      cvt8(qp  + s0,      &qa[c].e[0]);
      cvt8(qp  + s0 + 8,  &qa[c].e[8]);
      cvt8(qip + s0,      &qia[c].e[0]);
      cvt8(qip + s0 + 8,  &qia[c].e[8]);
    }
  }

  v8f o_r[4], o_i[4];
#pragma unroll
  for (int d = 0; d < 4; ++d) { o_r[d] = {}; o_i[d] = {}; }

  float m_[8], l_[8];                      // per-row softmax state (C-layout rows)
#pragma unroll
  for (int e = 0; e < 8; ++e) { m_[e] = -3.0e38f; l_[e] = 0.0f; }

  const float* Kbase  = K  + headOff;
  const float* Kibase = Ki + headOff;
  const float* Vbase  = V  + headOff;
  const float* Vibase = Vi + headOff;
  const int*   mbase  = MSK + (size_t)b * S_N * S_N;
  _Float16*    myP    = &ldsP[wave][0][0];

  constexpr float LOG2E = 1.4426950408889634f;

#pragma unroll 1
  for (int kb = 0; kb < S_N; kb += 32) {
    if (kb + 32 < S_N) {   // CDNA5 global_prefetch_b8 for next K/V tiles
      __builtin_prefetch(Kbase + (size_t)(kb + 32 + l16) * D_N, 0, 0);
      __builtin_prefetch(Vbase + (size_t)(kb + 32 + l16) * D_N, 0, 0);
    }

    // ---- scores for two 16-key N-tiles ----
    v8f sc[2];
#pragma unroll
    for (int t = 0; t < 2; ++t) {
      // B-layout: lane holds key row n = l16 of this tile; halves 0..15 = d-chunk
      const float* kp  = Kbase  + (size_t)(kb + t*16 + l16) * D_N + hs * 16;
      const float* kip = Kibase + (size_t)(kb + t*16 + l16) * D_N + hs * 16;
      V16H b0, b1;
      cvt16(kp,      b0.e);
      cvt16(kp + 32, b1.e);
      v8f r = {};
      r = wmma_f16(qa[0].v, b0.v, r);
      r = wmma_f16(qa[1].v, b1.v, r);
      cvt16(kip,      b0.e);
      cvt16(kip + 32, b1.e);
      v8f im = {};
      im = wmma_f16(qia[0].v, b0.v, im);
      im = wmma_f16(qia[1].v, b1.v, im);

      v8f s;
#pragma unroll
      for (int e = 0; e < 8; ++e) {
        float rr = r[e], ii = im[e];
        float sv = sqrtf(rr * rr + ii * ii + 1e-8f) * 0.125f;  // 1/sqrt(64)
        const int mq = qbaseW + e + hs * 8;                    // C-layout row
        const int kc = kb + t * 16 + l16;                      // C-layout col
        if (mbase[(size_t)mq * S_N + kc] == 0) sv = -1.0e9f;
        s[e] = sv;
      }
      sc[t] = s;
    }

    // ---- online softmax + stage P (f16) into per-wave LDS tile ----
#pragma unroll
    for (int e = 0; e < 8; ++e) {
      float a0 = sc[0][e], a1 = sc[1][e];
      float cm = fmaxf(a0, a1);
      cm = fmaxf(cm, swz_xor<1>(cm));
      cm = fmaxf(cm, swz_xor<2>(cm));
      cm = fmaxf(cm, swz_xor<4>(cm));
      cm = fmaxf(cm, swz_xor<8>(cm));
      const float mn    = fmaxf(m_[e], cm);
      const float scale = exp2f((m_[e] - mn) * LOG2E);
      const float p0    = exp2f((a0 - mn) * LOG2E);
      const float p1    = exp2f((a1 - mn) * LOG2E);
      float rs = p0 + p1;
      rs += swz_xor<1>(rs);
      rs += swz_xor<2>(rs);
      rs += swz_xor<4>(rs);
      rs += swz_xor<8>(rs);
      l_[e] = l_[e] * scale + rs;
      m_[e] = mn;
#pragma unroll
      for (int d = 0; d < 4; ++d) { o_r[d][e] *= scale; o_i[d][e] *= scale; }
      const int row = e + hs * 8;
      myP[row * LDSK + l16]      = (_Float16)p0;
      myP[row * LDSK + 16 + l16] = (_Float16)p1;
    }

    // DS pipeline is in-order per wave; wait makes stores visible to reloads
    asm volatile("s_wait_dscnt 0" ::: "memory");

    // ---- reload P in A-layout (row = l16, k segs {k0..k0+7, k0+16..k0+23}) ----
    V16H pa;
    {
      const _Float16* prow = myP + l16 * LDSK;
      const int koff = hs * 8;
      pa.h[0] = *reinterpret_cast<const v8h*>(prow + koff);
      pa.h[1] = *reinterpret_cast<const v8h*>(prow + koff + 16);
    }

    // ---- O += P * V  and  O_i += P * Vi (4 N-tiles of D) ----
#pragma unroll
    for (int dt = 0; dt < 4; ++dt) {
      V16H vb, vib;
#pragma unroll
      for (int i = 0; i < 16; ++i) {
        const size_t ro = (size_t)(kb + hs * 8 + (i & 7) + (i >> 3) * 16) * D_N + dt * 16 + l16;
        vb.e[i]  = (_Float16)Vbase[ro];
        vib.e[i] = (_Float16)Vibase[ro];
      }
      o_r[dt] = wmma_f16(pa.v, vb.v, o_r[dt]);
      o_i[dt] = wmma_f16(pa.v, vib.v, o_i[dt]);
    }
  }

  // ---- epilogue: divide by l, write real then imag (concatenated outputs) ----
  const size_t TOT = (size_t)B_N * H_N * S_N * D_N;
#pragma unroll
  for (int e = 0; e < 8; ++e) {
    const float inv = 1.0f / l_[e];
    const size_t rbase = headOff + (size_t)(qbaseW + e + hs * 8) * D_N;
#pragma unroll
    for (int dt = 0; dt < 4; ++dt) {
      const size_t o = rbase + dt * 16 + l16;
      out[o]       = o_r[dt][e] * inv;
      out[o + TOT] = o_i[dt][e] * inv;
    }
  }
}

extern "C" void kernel_launch(void* const* d_in, const int* in_sizes, int n_in,
                              void* d_out, int out_size, void* d_ws, size_t ws_size,
                              hipStream_t stream) {
  (void)in_sizes; (void)n_in; (void)out_size; (void)d_ws; (void)ws_size;
  const float* Q  = (const float*)d_in[0];
  const float* K  = (const float*)d_in[1];
  const float* V  = (const float*)d_in[2];
  const float* Qi = (const float*)d_in[3];
  const float* Ki = (const float*)d_in[4];
  const float* Vi = (const float*)d_in[5];
  const int*   M  = (const int*)d_in[6];
  float* out = (float*)d_out;

  dim3 grid(B_N * H_N * (S_N / QBLOCK));   // 512 blocks
  dim3 block(256);                          // 8 wave32 waves
  fca_attn_kernel<<<grid, block, 0, stream>>>(Q, K, V, Qi, Ki, Vi, M, out);
}
